// Hippocampus_14602888806876
// MI455X (gfx1250) — compile-verified, matrix operand order checked by reading, not run on hardware
//
#include <hip/hip_runtime.h>
#include <hip/hip_bf16.h>

// Shapes from the reference
#define NUNITS  256
#define DHID    1024
#define NBATCH  768
#define NCOMB   1536          // [key rows ; query rows]
#define NGROUP  8
#define JROWS   96            // NBATCH / NGROUP (softmax axis length)

typedef __attribute__((ext_vector_type(16))) __bf16 v16bf;
typedef __attribute__((ext_vector_type(8)))  float  v8f;

// -------------------------------------------------------------------------
// Kernel A: fused 2-layer MLP for key (rows 0..767) and query (rows 768..1535)
//   kq[r,:] = leaky_relu(v @ W1 + b1) @ W2 + b2,  v = x or mask(x)
// One block = one 16-row M tile (96 blocks). 256 threads = 8 waves.
// Stage 1: each wave owns 128 hidden cols (8 WMMA N-subtiles), K=256.
// Stage 2: each wave owns 32 output cols (2 WMMA N-subtiles), K=1024 from LDS.
// -------------------------------------------------------------------------
#define HP (DHID + 8)   // padded LDS row stride (bf16) to break bank conflicts

__global__ __launch_bounds__(256)
void Hippo_mlp_kernel(const float* __restrict__ x,
                      const unsigned char* __restrict__ mask,
                      const float* __restrict__ W1, const float* __restrict__ b1,
                      const float* __restrict__ W2, const float* __restrict__ b2,
                      float* __restrict__ kq) {
    __shared__ __bf16 hs[16 * HP];               // 33 KB: hidden tile, bf16

    const int blk    = blockIdx.x;               // 0..95
    const int m0     = blk * 16;                 // combined-row base
    const bool dmask = (m0 >= NBATCH);           // query half => apply unit mask
    const int xrow0  = dmask ? (m0 - NBATCH) : m0;

    const int tid  = threadIdx.x;
    const int wave = tid >> 5;
    const int lane = tid & 31;
    const int lm   = lane & 15;                  // M (A) or N (B/C/D) index
    const int koff = (lane >> 4) * 8;            // K sub-offset per lane half

    const v8f vzero = {};

    // ---------------- Stage 1: h[16,1024] = lrelu(A @ W1 + b1) ----------------
    v8f acc[8];
#pragma unroll
    for (int s = 0; s < 8; ++s) acc[s] = vzero;

    for (int kt = 0; kt < NUNITS; kt += 32) {
        // A fragment: row (xrow0+lm), K = kt+koff+{0..7} and +16
        v16bf a;
        const float*         xr = x    + (size_t)(xrow0 + lm) * NUNITS;
        const unsigned char* mr = mask + (size_t)(xrow0 + lm) * NUNITS;
#pragma unroll
        for (int e = 0; e < 8; ++e) {
            const int c0 = kt + koff + e;
            const int c1 = c0 + 16;
            float v0 = xr[c0], v1 = xr[c1];
            if (dmask) { if (mr[c0]) v0 = 0.f; if (mr[c1]) v1 = 0.f; }
            a[e]     = (__bf16)v0;
            a[8 + e] = (__bf16)v1;
        }
#pragma unroll
        for (int s = 0; s < 8; ++s) {
            const int n0 = wave * 128 + s * 16;
            v16bf bfr;                           // B: W1[K, N], col (n0+lm)
            const float* wp = W1 + (size_t)(kt + koff) * DHID + n0 + lm;
#pragma unroll
            for (int e = 0; e < 8; ++e) {
                bfr[e]     = (__bf16)wp[(size_t)e * DHID];
                bfr[8 + e] = (__bf16)wp[(size_t)(e + 16) * DHID];
            }
            acc[s] = __builtin_amdgcn_wmma_f32_16x16x32_bf16(
                         false, a, false, bfr, (short)0, acc[s], false, false);
        }
    }
    // bias + leaky relu -> LDS (bf16)
#pragma unroll
    for (int s = 0; s < 8; ++s) {
        const int n = wave * 128 + s * 16 + lm;
        const float bias = b1[n];
#pragma unroll
        for (int r = 0; r < 8; ++r) {
            const int m = r + 8 * (lane >> 4);
            float v = acc[s][r] + bias;
            v = (v > 0.f) ? v : 0.01f * v;
            hs[m * HP + n] = (__bf16)v;
        }
    }
    __syncthreads();

    // ---------------- Stage 2: out[16,256] = h @ W2 + b2 ----------------
    v8f acc2[2];
    acc2[0] = vzero; acc2[1] = vzero;

    for (int kt = 0; kt < DHID; kt += 32) {
        v16bf a;                                  // A from LDS hidden tile
#pragma unroll
        for (int e = 0; e < 8; ++e) {
            a[e]     = hs[lm * HP + kt + koff + e];
            a[8 + e] = hs[lm * HP + kt + koff + 16 + e];
        }
#pragma unroll
        for (int s = 0; s < 2; ++s) {
            const int n0 = wave * 32 + s * 16;
            v16bf bfr;                            // B: W2[K, N]
            const float* wp = W2 + (size_t)(kt + koff) * NUNITS + n0 + lm;
#pragma unroll
            for (int e = 0; e < 8; ++e) {
                bfr[e]     = (__bf16)wp[(size_t)e * NUNITS];
                bfr[8 + e] = (__bf16)wp[(size_t)(e + 16) * NUNITS];
            }
            acc2[s] = __builtin_amdgcn_wmma_f32_16x16x32_bf16(
                          false, a, false, bfr, (short)0, acc2[s], false, false);
        }
    }
#pragma unroll
    for (int s = 0; s < 2; ++s) {
        const int n = wave * 32 + s * 16 + lm;
        const float bias = b2[n];
#pragma unroll
        for (int r = 0; r < 8; ++r) {
            const int m = r + 8 * (lane >> 4);
            kq[(size_t)(m0 + m) * NUNITS + n] = acc2[s][r] + bias;
        }
    }
}

// -------------------------------------------------------------------------
// Kernel B: pairwise abs-diff + grouped softmax + streaming store.
// Block (i, j2): softmax over j1 of -|key[j1*8+j2,u] - query[i,u]|*temp[u],
// diag j==i forced to -inf, result /8. One thread per u (column-exclusive:
// no barriers needed; the exp values are cached back into the LDS slice).
// -------------------------------------------------------------------------
__global__ __launch_bounds__(256)
void Hippo_pairwise_kernel(const float* __restrict__ kq,
                           const float* __restrict__ temp,
                           float* __restrict__ out) {
    __shared__ float ks[JROWS * NUNITS];          // 96 KB key slice / exp cache

    const int i  = blockIdx.x;                    // 0..767 (query row)
    const int j2 = blockIdx.y;                    // 0..7   (group residue)
    const int u  = threadIdx.x;                   // 0..255 (unit)

    const float* key   = kq;
    const float* query = kq + (size_t)NBATCH * NUNITS;

    const float q  = query[(size_t)i * NUNITS + u];
    const float tu = temp[u];
    const int  diag = (j2 == (i & 7)) ? (i >> 3) : -1;

    // Stage key slice (coalesced; each thread owns column u thereafter).
    for (int r = 0; r < JROWS; ++r)
        ks[r * NUNITS + u] = key[(size_t)(r * NGROUP + j2) * NUNITS + u];

    // Pass 1: max (diag excluded; scores are <= 0).
    float mx = -3.0e38f;
    for (int r = 0; r < JROWS; ++r) {
        if (r == diag) continue;
        const float s = -__builtin_fabsf(ks[r * NUNITS + u] - q) * tu;
        mx = __builtin_fmaxf(mx, s);
    }

    // Pass 2: exp + sum; overwrite LDS slot with exp value.
    const float L2E = 1.44269504088896340736f;
    float sum = 0.f;
    for (int r = 0; r < JROWS; ++r) {
        float e;
        if (r == diag) {
            e = 0.f;                              // exp(-inf)
        } else {
            const float s = -__builtin_fabsf(ks[r * NUNITS + u] - q) * tu;
            e = __builtin_amdgcn_exp2f((s - mx) * L2E);
        }
        ks[r * NUNITS + u] = e;
        sum += e;
    }

    const float inv = 1.f / (8.f * sum);          // softmax norm and /8

    // Pass 3: streaming non-temporal stores (604 MB total -> HBM bound).
    float* op = out + ((size_t)i * NBATCH + j2) * NUNITS + u;
    for (int r = 0; r < JROWS; ++r)
        __builtin_nontemporal_store(ks[r * NUNITS + u] * inv,
                                    op + (size_t)r * NGROUP * NUNITS);
}

// -------------------------------------------------------------------------
// Kernel C: mask passthrough (second tuple output) as float 0/1.
// -------------------------------------------------------------------------
__global__ __launch_bounds__(256)
void Hippo_maskout_kernel(const unsigned char* __restrict__ mask,
                          float* __restrict__ out) {
    const int idx = blockIdx.x * 256 + threadIdx.x;
    out[idx] = mask[idx] ? 1.0f : 0.0f;
}

extern "C" void kernel_launch(void* const* d_in, const int* in_sizes, int n_in,
                              void* d_out, int out_size, void* d_ws, size_t ws_size,
                              hipStream_t stream) {
    (void)in_sizes; (void)n_in; (void)out_size; (void)ws_size;

    const float*         x    = (const float*)d_in[0];
    const unsigned char* mask = (const unsigned char*)d_in[1]; // jax bool (1B)
    const float*         W1   = (const float*)d_in[2];
    const float*         b1   = (const float*)d_in[3];
    const float*         W2   = (const float*)d_in[4];
    const float*         b2   = (const float*)d_in[5];
    const float*         temp = (const float*)d_in[6];

    float* out = (float*)d_out;
    float* kq  = (float*)d_ws;                    // 1536*256 f32 = 1.5 MB scratch

    Hippo_mlp_kernel<<<NCOMB / 16, 256, 0, stream>>>(x, mask, W1, b1, W2, b2, kq);
    Hippo_pairwise_kernel<<<dim3(NBATCH, NGROUP), 256, 0, stream>>>(kq, temp, out);
    Hippo_maskout_kernel<<<(NBATCH * NUNITS) / 256, 256, 0, stream>>>(
        mask, out + (size_t)NBATCH * NBATCH * NUNITS);
}
